// Attn_spa_46823733461712
// MI455X (gfx1250) — matrix operand order repaired and checked by reading, not compiled, not run on hardware
//
#include <hip/hip_runtime.h>
#include <hip/hip_bf16.h>

typedef _Float16 f16;
typedef __attribute__((ext_vector_type(8)))  _Float16 v8h;
typedef __attribute__((ext_vector_type(16))) _Float16 v16h;
typedef __attribute__((ext_vector_type(8)))  float    v8f;

#define DIM 1024
#define HEADS 16
#define DH 64
#define BATCH 4
#define SEQ 1024
#define MTOT (BATCH * SEQ)   // 4096

enum { EP_F16 = 0, EP_SILU = 1, EP_SCALE = 2, EP_VT = 3, EP_FINAL = 4 };

__device__ __forceinline__ v8f wmma_f16(v16h a, v16h b, v8f c) {
  return __builtin_amdgcn_wmma_f32_16x16x32_f16(false, a, false, b, (short)0, c,
                                                false, false);
}

// A-fragment (16x32 f16): lane L -> row L%16, half h=L/16.
// elems 0..7  = K[k0 + h*8 + i], elems 8..15 = K[k0 + 16 + h*8 + i]
__device__ __forceinline__ v16h load_a_frag(const f16* base, int stride, int k0) {
  int lane = threadIdx.x & 31;
  const f16* p = base + (size_t)(lane & 15) * stride + k0 + ((lane >> 4) << 3);
  v8h lo = *(const v8h*)p;
  v8h hi = *(const v8h*)(p + 16);
  v16h a;
#pragma unroll
  for (int i = 0; i < 8; ++i) { a[i] = lo[i]; a[i + 8] = hi[i]; }
  return a;
}

// B-fragment (32x16 f16) from B^T rows: lane L -> col L%16, half h=L/16.
// elems 0..7 = Bt[col][k0 + h*8 + i], elems 8..15 = Bt[col][k0 + 16 + h*8 + i]
__device__ __forceinline__ v16h load_b_frag(const f16* base, int stride, int k0) {
  int lane = threadIdx.x & 31;
  const f16* p = base + (size_t)(lane & 15) * stride + k0 + ((lane >> 4) << 3);
  v8h lo = *(const v8h*)p;
  v8h hi = *(const v8h*)(p + 16);
  v16h b;
#pragma unroll
  for (int i = 0; i < 8; ++i) { b[i] = lo[i]; b[i + 8] = hi[i]; }
  return b;
}

// ---------------- elementwise f32 -> f16 ----------------
__global__ void cvt_f32_f16(const float* __restrict__ in, f16* __restrict__ out,
                            int n) {
  int i = blockIdx.x * blockDim.x + threadIdx.x;
  if (i < n) out[i] = (f16)in[i];
}

// ---------------- W [in][out] f32 -> Wt [out][in] f16 ----------------
__global__ void transpose_cvt(const float* __restrict__ W, f16* __restrict__ Wt,
                              int n) {
  __shared__ float tile[32][33];
  int x = blockIdx.x * 32 + threadIdx.x;
  int y0 = blockIdx.y * 32;
#pragma unroll
  for (int j = threadIdx.y; j < 32; j += 8)
    tile[j][threadIdx.x] = W[(size_t)(y0 + j) * n + x];
  __syncthreads();
  int ox = blockIdx.y * 32 + threadIdx.x;   // = original row
  int oy0 = blockIdx.x * 32;                // = original col
#pragma unroll
  for (int j = threadIdx.y; j < 32; j += 8)
    Wt[(size_t)(oy0 + j) * n + ox] = (f16)tile[threadIdx.x][j];
}

// ---------------- generic 128x128 WMMA GEMM:  C = A @ Bt^T ----------------
// A: [M][K] f16 row-major (lda), Bt: [N][K] f16 row-major (ldb)
__global__ void __launch_bounds__(256) gemm_wmma(
    const f16* __restrict__ A, size_t sAb, int lda,
    const f16* __restrict__ Bt, size_t sBb, int ldb,
    void* __restrict__ Cout, size_t sCb,
    const float* __restrict__ bias, const float* __restrict__ residual,
    int N, int K, int mode, float scale) {
  __shared__ f16 As[128][40];
  __shared__ f16 Bs[128][40];

  const int tid = threadIdx.x;
  const int lane = tid & 31;
  const int w = tid >> 5;                 // 8 waves
  const int m0 = blockIdx.y * 128;
  const int n0 = blockIdx.x * 128;
  const size_t batch = blockIdx.z;
  A += batch * sAb;
  Bt += batch * sBb;

  v8f acc[8];
#pragma unroll
  for (int nt = 0; nt < 8; ++nt)
#pragma unroll
    for (int r = 0; r < 8; ++r) acc[nt][r] = 0.f;

  const int lrow = tid >> 1;              // 0..127
  const int lseg = (tid & 1) * 16;        // halves

  for (int kb = 0; kb < K; kb += 32) {
    __syncthreads();
    const uint4* sa = (const uint4*)(A + (size_t)(m0 + lrow) * lda + kb + lseg);
    uint4* da = (uint4*)&As[lrow][lseg];
    da[0] = sa[0]; da[1] = sa[1];
    const uint4* sb = (const uint4*)(Bt + (size_t)(n0 + lrow) * ldb + kb + lseg);
    uint4* db = (uint4*)&Bs[lrow][lseg];
    db[0] = sb[0]; db[1] = sb[1];
    __syncthreads();

    v16h af = load_a_frag(&As[w * 16][0], 40, 0);
#pragma unroll
    for (int nt = 0; nt < 8; ++nt) {
      v16h bf = load_b_frag(&Bs[nt * 16][0], 40, 0);
      acc[nt] = wmma_f16(af, bf, acc[nt]);
    }
  }

  // epilogue (C layout: VGPR r -> row r + 8*(lane/16); col = lane%16)
  const int hh = lane >> 4;
  const int cl = lane & 15;
  f16* C16 = (f16*)Cout + batch * sCb;
#pragma unroll
  for (int nt = 0; nt < 8; ++nt) {
#pragma unroll
    for (int r = 0; r < 8; ++r) {
      int row = m0 + w * 16 + r + 8 * hh;
      int col = n0 + nt * 16 + cl;
      float val = acc[nt][r];
      if (mode == EP_F16) {
        C16[(size_t)row * N + col] = (f16)val;
      } else if (mode == EP_SILU) {
        val += bias[col];
        val = val / (1.f + __expf(-val));
        C16[(size_t)row * N + col] = (f16)val;
      } else if (mode == EP_SCALE) {
        C16[(size_t)row * N + col] = (f16)(val * scale);
      } else if (mode == EP_VT) {
        // v [4096][1024] -> vt [B][H][Dh][N]
        int bb = row >> 10, nseq = row & 1023;
        int hx = col >> 6, dh = col & 63;
        C16[((size_t)(bb * HEADS + hx) * DH + dh) * SEQ + nseq] = (f16)val;
      } else {  // EP_FINAL: f32 + bias + residual
        ((float*)Cout)[(size_t)row * N + col] =
            val + bias[col] + residual[(size_t)row * N + col];
      }
    }
  }
}

// ---------------- fused flash attention with logit bias ----------------
// block: (b, h, 128 q rows); 8 waves x 16 rows each; loop over 64-col k chunks
__global__ void __launch_bounds__(256) attn_kernel(
    const f16* __restrict__ q, const f16* __restrict__ k,
    const f16* __restrict__ vt, const f16* __restrict__ logits,
    const float* __restrict__ pi_ptr, f16* __restrict__ out) {
  __shared__ f16 qs[128][72];
  __shared__ f16 ks[64][72];
  __shared__ f16 vs[64][72];        // vt tile: [dh][k]
  __shared__ f16 ps[8][16][72];     // per-wave: logits tile, then P tile

  const int tid = threadIdx.x;
  const int lane = tid & 31;
  const int w = tid >> 5;
  const int hh = lane >> 4;
  const int cl = lane & 15;
  const int b = blockIdx.z;
  const int h = blockIdx.y;
  const int q0 = blockIdx.x * 128;

  const f16* qg = q + (size_t)b * SEQ * DIM + h * DH;
  const f16* kg = k + (size_t)b * SEQ * DIM + h * DH;
  const f16* vg = vt + ((size_t)b * HEADS + h) * DH * SEQ;
  const f16* lg = logits + (size_t)b * SEQ * SEQ;
  const float pi = pi_ptr[0];

  {  // q tile: 128 rows x 64 halves
    int row = tid >> 1, seg = (tid & 1) * 32;
    const uint4* s = (const uint4*)(qg + (size_t)(q0 + row) * DIM + seg);
    uint4* d = (uint4*)&qs[row][seg];
    d[0] = s[0]; d[1] = s[1]; d[2] = s[2]; d[3] = s[3];
  }

  float mrow[8], lrow[8];
  v8f oacc[4];
#pragma unroll
  for (int r = 0; r < 8; ++r) { mrow[r] = -1e30f; lrow[r] = 0.f; }
#pragma unroll
  for (int dt = 0; dt < 4; ++dt)
#pragma unroll
    for (int r = 0; r < 8; ++r) oacc[dt][r] = 0.f;

  for (int kc = 0; kc < SEQ; kc += 64) {
    __syncthreads();
    {  // k tile [64 x 64] and vt tile [64 x 64]
      int row = tid >> 2, seg = (tid & 3) * 16;
      const uint4* s = (const uint4*)(kg + (size_t)(kc + row) * DIM + seg);
      uint4* d = (uint4*)&ks[row][seg];
      d[0] = s[0]; d[1] = s[1];
      const uint4* s2 = (const uint4*)(vg + (size_t)row * SEQ + kc + seg);
      uint4* d2 = (uint4*)&vs[row][seg];
      d2[0] = s2[0]; d2[1] = s2[1];
    }
    __syncthreads();
    {  // logits tile for this wave's 16 rows (into ps[w])
      int row = lane >> 1, seg = (lane & 1) * 32;
      const uint4* s =
          (const uint4*)(lg + (size_t)(q0 + w * 16 + row) * SEQ + kc + seg);
      uint4* d = (uint4*)&ps[w][row][seg];
      d[0] = s[0]; d[1] = s[1]; d[2] = s[2]; d[3] = s[3];
    }

    // S[16x64] = q_tile @ k_tile^T  (K = 64 -> two k-steps)
    v16h a0 = load_a_frag(&qs[w * 16][0], 72, 0);
    v16h a1 = load_a_frag(&qs[w * 16][0], 72, 32);
    v8f s4[4];
#pragma unroll
    for (int nt = 0; nt < 4; ++nt) {
#pragma unroll
      for (int r = 0; r < 8; ++r) s4[nt][r] = 0.f;
      v16h b0 = load_b_frag(&ks[nt * 16][0], 72, 0);
      s4[nt] = wmma_f16(a0, b0, s4[nt]);
      v16h b1 = load_b_frag(&ks[nt * 16][0], 72, 32);
      s4[nt] = wmma_f16(a1, b1, s4[nt]);
    }

    // final_logits = S/8 + pi * logits
#pragma unroll
    for (int nt = 0; nt < 4; ++nt)
#pragma unroll
      for (int r = 0; r < 8; ++r) {
        float lv = (float)ps[w][r + 8 * hh][nt * 16 + cl];
        s4[nt][r] = s4[nt][r] * 0.125f + pi * lv;
      }

    // row max over this chunk (16 lanes per row -> xor reduce within half)
    float rm[8];
#pragma unroll
    for (int r = 0; r < 8; ++r) {
      float v = fmaxf(fmaxf(s4[0][r], s4[1][r]), fmaxf(s4[2][r], s4[3][r]));
      v = fmaxf(v, __shfl_xor(v, 1));
      v = fmaxf(v, __shfl_xor(v, 2));
      v = fmaxf(v, __shfl_xor(v, 4));
      v = fmaxf(v, __shfl_xor(v, 8));
      rm[r] = v;
    }

    // online softmax rescale
#pragma unroll
    for (int r = 0; r < 8; ++r) {
      float mn = fmaxf(mrow[r], rm[r]);
      float alpha = __expf(mrow[r] - mn);
      mrow[r] = mn;
      lrow[r] *= alpha;
#pragma unroll
      for (int dt = 0; dt < 4; ++dt) oacc[dt][r] *= alpha;
    }

    // P = exp(S - m); stash in ps[w] (C-layout -> A-layout via LDS)
#pragma unroll
    for (int nt = 0; nt < 4; ++nt)
#pragma unroll
      for (int r = 0; r < 8; ++r) {
        float p = __expf(s4[nt][r] - mrow[r]);
        lrow[r] += p;  // per-lane partial; reduced at the end
        ps[w][r + 8 * hh][nt * 16 + cl] = (f16)p;
      }

    // O += P @ V   (A = ps[w] 16x64, Bt = vs [dh][k])
    v16h p0 = load_a_frag(&ps[w][0][0], 72, 0);
    v16h p1 = load_a_frag(&ps[w][0][0], 72, 32);
#pragma unroll
    for (int dt = 0; dt < 4; ++dt) {
      v16h vb0 = load_b_frag(&vs[dt * 16][0], 72, 0);
      oacc[dt] = wmma_f16(p0, vb0, oacc[dt]);
      v16h vb1 = load_b_frag(&vs[dt * 16][0], 72, 32);
      oacc[dt] = wmma_f16(p1, vb1, oacc[dt]);
    }
  }

  // finalize: reduce row sums across the 16-lane group, divide, store
#pragma unroll
  for (int r = 0; r < 8; ++r) {
    float l = lrow[r];
    l += __shfl_xor(l, 1);
    l += __shfl_xor(l, 2);
    l += __shfl_xor(l, 4);
    l += __shfl_xor(l, 8);
    lrow[r] = 1.f / l;
  }
#pragma unroll
  for (int dt = 0; dt < 4; ++dt)
#pragma unroll
    for (int r = 0; r < 8; ++r) {
      int row = q0 + w * 16 + r + 8 * hh;
      int col = h * DH + dt * 16 + cl;
      out[((size_t)b * SEQ + row) * DIM + col] = (f16)(oacc[dt][r] * lrow[r]);
    }
}

// ---------------- host ----------------
extern "C" void kernel_launch(void* const* d_in, const int* in_sizes, int n_in,
                              void* d_out, int out_size, void* d_ws,
                              size_t ws_size, hipStream_t stream) {
  const float* x     = (const float*)d_in[0];
  const float* Wq    = (const float*)d_in[1];
  const float* Wk    = (const float*)d_in[2];
  const float* Wv    = (const float*)d_in[3];
  const float* Wproj = (const float*)d_in[4];
  const float* bproj = (const float*)d_in[5];
  const float* Wpre  = (const float*)d_in[6];
  const float* bpre  = (const float*)d_in[7];
  const float* pi    = (const float*)d_in[8];

  // workspace layout (f16 elements; ~69 MB total)
  f16* p = (f16*)d_ws;
  f16* xh     = p; p += (size_t)MTOT * DIM;
  f16* WqT    = p; p += (size_t)DIM * DIM;
  f16* WkT    = p; p += (size_t)DIM * DIM;
  f16* WvT    = p; p += (size_t)DIM * DIM;
  f16* WpreT  = p; p += (size_t)DIM * DIM;
  f16* WprojT = p; p += (size_t)DIM * DIM;
  f16* pre    = p; p += (size_t)MTOT * DIM;
  f16* qb     = p; p += (size_t)MTOT * DIM;
  f16* kb     = p; p += (size_t)MTOT * DIM;
  f16* vtb    = p; p += (size_t)MTOT * DIM;      // [B][H][Dh][N]
  f16* lgt    = p; p += (size_t)BATCH * SEQ * SEQ;
  f16* ao     = p; p += (size_t)MTOT * DIM;

  dim3 tb(256);

  // convert x, transpose+convert weights
  cvt_f32_f16<<<(MTOT * DIM) / 256, tb, 0, stream>>>(x, xh, MTOT * DIM);
  dim3 tg(32, 32), tbt(32, 8);
  transpose_cvt<<<tg, tbt, 0, stream>>>(Wq, WqT, DIM);
  transpose_cvt<<<tg, tbt, 0, stream>>>(Wk, WkT, DIM);
  transpose_cvt<<<tg, tbt, 0, stream>>>(Wv, WvT, DIM);
  transpose_cvt<<<tg, tbt, 0, stream>>>(Wpre, WpreT, DIM);
  transpose_cvt<<<tg, tbt, 0, stream>>>(Wproj, WprojT, DIM);

  dim3 gFull(DIM / 128, MTOT / 128, 1);
  // pre = silu(x @ Wpre + bpre)
  gemm_wmma<<<gFull, tb, 0, stream>>>(xh, 0, DIM, WpreT, 0, DIM, pre, 0, bpre,
                                      nullptr, DIM, DIM, EP_SILU, 1.f);
  // q, k, vt = x @ {Wq, Wk, Wv}
  gemm_wmma<<<gFull, tb, 0, stream>>>(xh, 0, DIM, WqT, 0, DIM, qb, 0, nullptr,
                                      nullptr, DIM, DIM, EP_F16, 1.f);
  gemm_wmma<<<gFull, tb, 0, stream>>>(xh, 0, DIM, WkT, 0, DIM, kb, 0, nullptr,
                                      nullptr, DIM, DIM, EP_F16, 1.f);
  gemm_wmma<<<gFull, tb, 0, stream>>>(xh, 0, DIM, WvT, 0, DIM, vtb, 0, nullptr,
                                      nullptr, DIM, DIM, EP_VT, 1.f);
  // logits = (pre @ pre^T) / 32, batched over B
  dim3 gLog(SEQ / 128, SEQ / 128, BATCH);
  gemm_wmma<<<gLog, tb, 0, stream>>>(pre, (size_t)SEQ * DIM, DIM, pre,
                                     (size_t)SEQ * DIM, DIM, lgt,
                                     (size_t)SEQ * SEQ, nullptr, nullptr, SEQ,
                                     DIM, EP_SCALE, 1.f / 32.f);
  // fused softmax(pi*logits + qk^T/8) @ v
  dim3 gAttn(SEQ / 128, HEADS, BATCH);
  attn_kernel<<<gAttn, tb, 0, stream>>>(qb, kb, vtb, lgt, pi, ao);
  // out = ao @ Wproj + bproj + x
  gemm_wmma<<<gFull, tb, 0, stream>>>(ao, 0, DIM, WprojT, 0, DIM, d_out, 0,
                                      bproj, x, DIM, DIM, EP_FINAL, 1.f);
}
